// Poly_Kernel_62749472195000
// MI455X (gfx1250) — matrix operand order repaired and checked by reading, not compiled, not run on hardware
//
#include <hip/hip_runtime.h>
#include <hip/hip_bf16.h>

// CDNA5 / gfx1250, wave32. bf16 WMMA (f32 accumulate) im2col conv3d,
// x-slab staged to LDS via the Tensor Data Mover (TENSOR_LOAD_TO_LDS).
typedef __attribute__((ext_vector_type(16))) __bf16 v16bf;
typedef __attribute__((ext_vector_type(8)))  float  v8f;
typedef __attribute__((ext_vector_type(4)))  unsigned int u32x4;
typedef __attribute__((ext_vector_type(8)))  int  i32x8;
typedef __attribute__((ext_vector_type(4)))  int  i32x4;

union BFrag {
    v16bf v;
    __bf16 e[16];
};

// Shapes (hard-wired to the reference problem)
//   x: (B=8, C=16, D=64, H=64, W=64) f32
//   w: (O=32, K=128) f32, k = c*8 + kd*4 + kh*2 + kw
//   out: (8, 32, 63, 63, 63) f32, out = (conv + 1)^2
#define CIN   16
#define OCH   32
#define DIMW  64
#define OUTW  63
#define KDIM  128

#if defined(__gfx1250__) && __has_builtin(__builtin_amdgcn_tensor_load_to_lds)
#define USE_TDM 1
#else
#define USE_TDM 0
#endif

__global__ __launch_bounds__(128) void conv3d_k2_wmma_bf16(
    const float* __restrict__ x,
    const float* __restrict__ w,
    float* __restrict__ out)
{
    // Block identity: one (b, dz, hz) output row of 63 wz positions.
    const int hz = blockIdx.x;     // 0..62
    const int dz = blockIdx.y;     // 0..62
    const int b  = blockIdx.z;     // 0..7

    const int t    = threadIdx.x;  // 0..127 (4 waves)
    const int lane = t & 31;
    const int wv   = t >> 5;       // wave id 0..3 -> wz tile base 16*wv
    const int half = lane >> 4;    // lane half (0: lanes 0-15, 1: lanes 16-31)
    const int nn   = lane & 15;    // N index inside 16x16 tile

    // LDS: f32 input slab, rows [c*4 + kd*2 + kh][w], w = 0..63 (+overread pad).
    // TDM fills it contiguously: x fastest (w), then kh, kd, c == this layout.
    __shared__ __align__(16) float ldsXf[64 * 64 + 64];
    // LDS: weights pre-swizzled into WMMA A-fragment order:
    //   aFrag[((ck*2 + mh)*32 + lane)*16 + i]  (ck = K chunk, mh = M half 0/1)
    __shared__ __align__(32) __bf16 aFrag[4 * 2 * 32 * 16];

    // ---------------- Phase B issue: x slab -> LDS via Tensor Data Mover ------
    const float* slab = x + ((((long)b * CIN) * DIMW + dz) * DIMW + hz) * DIMW;
#if USE_TDM
    if (t == 0) {
        const unsigned long long ga = (unsigned long long)(const void*)slab;
        const unsigned lba = (unsigned)(unsigned long long)(const void*)&ldsXf[0];
        // D# group 0: count=1 (valid), lds_addr, global_addr[56:0], type=2.
        u32x4 g0 = { 1u,
                     lba,
                     (unsigned)(ga & 0xffffffffull),
                     (unsigned)((ga >> 32) & 0x01ffffffull) | (2u << 30) };
        // D# group 1: data_size=4B; tensor_dim0=64 (W), tensor_dim1=64 (H);
        // tile_dim0=64 (w), tile_dim1=2 (kh), tile_dim2=2 (kd);
        // dim0_stride=64 (kh step), dim1_stride=4096 (kd step).
        i32x8 g1 = { (int)(2u << 16),                 // wg_mask=0, data_size=2 (4B)
                     (int)(64u << 16),                // atomic_barrier=0 | dim0.lo16<<16
                     (int)(0u | (64u << 16)),         // dim0.hi16 | dim1.lo16<<16
                     (int)(0u | (64u << 16)),         // dim1.hi16 | tile_dim0<<16
                     (int)(2u | (2u << 16)),          // tile_dim1 | tile_dim2<<16
                     (int)64,                         // dim0_stride[31:0]
                     (int)(0u | (4096u << 16)),       // dim0_stride[47:32] | dim1_stride[15:0]<<16
                     (int)0 };                        // dim1_stride[47:16]
        // D# group 2: tensor_dim2=64 (D), tensor_dim3=16 (C),
        // dim2_stride=262144 (c step), tile_dim3=16 (c).
        i32x4 g2 = { 64, 16, 262144, (int)(0u | (16u << 16)) };
        // D# group 3: no 5th dimension.
        i32x4 g3 = { 0, 0, 0, 0 };
        // clang-23 lane: 6-arg form (g0, g1, g2, g3, <extra v8i>, cpol).
        i32x8 g4 = { 0, 0, 0, 0, 0, 0, 0, 0 };
        __builtin_amdgcn_tensor_load_to_lds(g0, g1, g2, g3, g4, 0);
    }
#else
    // Fallback: manual staging. 64 rows x 64 floats, 2 threads/row, 8 float4 each.
    {
        const int r    = t >> 1;
        const int part = t & 1;
        const float* src = slab + ((long)(r >> 2) * DIMW * DIMW * DIMW) // c
                                + ((long)((r >> 1) & 1) * DIMW * DIMW)  // kd
                                + ((long)(r & 1) * DIMW);               // kh
        #pragma unroll
        for (int q = 0; q < 8; ++q) {
            const int wc = part * 32 + q * 4;
            *reinterpret_cast<float4*>(&ldsXf[r * 64 + wc]) =
                *reinterpret_cast<const float4*>(src + wc);
        }
    }
#endif

    // ---------------- Phase A: weights -> A fragments (bf16, lane-swizzled) ----
    // Overlaps with the in-flight TDM DMA. 256 (ck, mh, lane) triples,
    // 16 bf16 each; 2 triples per thread.
    #pragma unroll
    for (int s = 0; s < 2; ++s) {
        const int T   = t * 2 + s;          // 0..255
        const int ck  = T >> 6;             // K chunk 0..3
        const int mh  = (T >> 5) & 1;       // M half
        const int ln  = T & 31;             // destination lane
        const int lh  = ln >> 4;            // lane half
        const int mrow = mh * 16 + (ln & 15);
        const int base = ((ck * 2 + mh) * 32 + ln) * 16;
        #pragma unroll
        for (int j = 0; j < 8; ++j) {
            const int i = 2 * j;
            // Contiguous k order used here (B matches): K = 16*half + i, pairs contiguous.
            const int k = ck * 32 + lh * 16 + i;
            const float2 wv2 = *reinterpret_cast<const float2*>(w + mrow * KDIM + k);
            aFrag[base + i]     = (__bf16)wv2.x;
            aFrag[base + i + 1] = (__bf16)wv2.y;
        }
    }

#if USE_TDM
    // Drain the TDM transfer. Per-wave counter: no-op for waves 1-3, and the
    // barrier below publishes wave 0's LDS contents to the whole workgroup.
#if __has_builtin(__builtin_amdgcn_s_wait_tensorcnt)
    __builtin_amdgcn_s_wait_tensorcnt(0);
#else
    asm volatile("s_wait_tensorcnt 0x0" ::: "memory");
#endif
#endif
    __syncthreads();

    // ---------------- Phase C: WMMA K-loop ------------------------------------
    const int w0 = wv * 16;          // wz tile base for this wave
    const int wn = w0 + nn;          // this lane's wz column (kw=0 tap)

    v8f acc0 = {};   // O rows 0..15
    v8f acc1 = {};   // O rows 16..31

    #pragma unroll
    for (int ck = 0; ck < 4; ++ck) {
        // A fragments: two ds_load_b128 each (32B contiguous per lane).
        const v16bf aLo = *reinterpret_cast<const v16bf*>(&aFrag[((ck * 2 + 0) * 32 + lane) * 16]);
        const v16bf aHi = *reinterpret_cast<const v16bf*>(&aFrag[((ck * 2 + 1) * 32 + lane) * 16]);

        // B fragment: contiguous k order matching A -> lane half h holds K = 16h..16h+15.
        BFrag bf;
        #pragma unroll
        for (int i = 0; i < 16; i += 2) {
            const int k   = ck * 32 + half * 16 + i;  // even -> kw=0
            const int c   = k >> 3;
            const int off = k & 7;                     // kd*4 + kh*2 + kw
            const int row = c * 4 + (off >> 1);
            bf.e[i]     = (__bf16)ldsXf[row * 64 + wn];     // kw = 0
            bf.e[i + 1] = (__bf16)ldsXf[row * 64 + wn + 1]; // kw = 1
        }

        acc0 = __builtin_amdgcn_wmma_f32_16x16x32_bf16(
            false, aLo, false, bf.v, (short)0, acc0, false, false);
        acc1 = __builtin_amdgcn_wmma_f32_16x16x32_bf16(
            false, aHi, false, bf.v, (short)0, acc1, false, false);
    }

    // ---------------- Epilogue: (y+1)^2, masked store -------------------------
    if (wn < OUTW) {
        #pragma unroll
        for (int r = 0; r < 8; ++r) {
            const int m = r + half * 8;   // D-matrix layout: VGPR r, lane half -> M
            float v0 = acc0[r] + 1.0f; v0 *= v0;
            float v1 = acc1[r] + 1.0f; v1 *= v1;
            out[((((long)b * OCH + m)      * OUTW + dz) * OUTW + hz) * OUTW + wn] = v0;
            out[((((long)b * OCH + m + 16) * OUTW + dz) * OUTW + hz) * OUTW + wn] = v1;
        }
    }
}

extern "C" void kernel_launch(void* const* d_in, const int* in_sizes, int n_in,
                              void* d_out, int out_size, void* d_ws, size_t ws_size,
                              hipStream_t stream) {
    const float* x = (const float*)d_in[0];
    const float* w = (const float*)d_in[1];
    float* out = (float*)d_out;
    (void)in_sizes; (void)n_in; (void)out_size; (void)d_ws; (void)ws_size;

    dim3 grid(OUTW, OUTW, 8);   // (hz, dz, b)
    dim3 block(128);            // 4 waves (wave32)
    conv3d_k2_wmma_bf16<<<grid, block, 0, stream>>>(x, w, out);
}
